// VQModule_6966436954592
// MI455X (gfx1250) — matrix operand order repaired and checked by reading, not compiled, not run on hardware
//
#include <hip/hip_runtime.h>
#include <hip/hip_bf16.h>

typedef __attribute__((ext_vector_type(16))) _Float16 v16h;
typedef __attribute__((ext_vector_type(8)))  _Float16 v8h;
typedef __attribute__((ext_vector_type(8)))  float    v8f;

#define EMB   64
#define DICT  512
#define NB    32
#define NHW   64
// N rows = 32*64*64 = 131072, tiles of 16 rows -> 8192 tiles, 8 waves/block -> 1024 blocks

// LDS carve: f16 codebook (512*64*2 = 65536 B) | 0.5*||e||^2 (512*4 = 2048 B) | ids (8*16*4 = 512 B)
#define LDS_CB_BYTES   65536
#define LDS_NORM_BYTES 2048
#define LDS_IDS_BYTES  512
#define LDS_TOTAL      (LDS_CB_BYTES + LDS_NORM_BYTES + LDS_IDS_BYTES)

__global__ void vq_zero(float* ws, int n) {
    int i = blockIdx.x * blockDim.x + threadIdx.x;
    if (i < n) ws[i] = 0.0f;
}

__global__ void vq_main(const float* __restrict__ inp,      // (32, 64, 64, 64)
                        const float* __restrict__ embed,    // (512, 64)
                        float* __restrict__ q_out,          // (32, 64, 64, 64)
                        float* __restrict__ ids_out,        // (131072,) as float
                        float* __restrict__ ws_hist,        // (512,)
                        float* __restrict__ ws_esum,        // (64, 512)
                        float* __restrict__ ws_commit)      // (1,)
{
    extern __shared__ char smem[];
    _Float16* cb  = (_Float16*)smem;
    float*    nrm = (float*)(smem + LDS_CB_BYTES);
    int*      idsh= (int*)(smem + LDS_CB_BYTES + LDS_NORM_BYTES);

    const int tid = threadIdx.x;

    // ---- Stage codebook to LDS as f16; exact 0.5*||e||^2 from f32 ----
    for (int i = tid; i < DICT * EMB; i += 256)
        cb[i] = (_Float16)embed[i];
    for (int d = tid; d < DICT; d += 256) {
        const float* e = embed + d * EMB;
        float s = 0.0f;
        #pragma unroll 8
        for (int k = 0; k < EMB; ++k) { float v = e[k]; s += v * v; }
        nrm[d] = 0.5f * s;
    }
    __syncthreads();

    const int wave = tid >> 5;
    const int lane = tid & 31;
    const int g    = lane >> 4;     // half-wave (K-group for A/B layout)
    const int col  = lane & 15;     // M row for A fill; N column for B/C

    // Tile T: rows n = ((bb*64 + ww0+m)*64 + hh), m = 0..15  (flatten order is (b, w, h))
    const int T   = blockIdx.x * 8 + wave;          // [0, 8192)
    const int ww0 = (T & 3) << 4;
    const int hh  = (T >> 2) & 63;
    const int bb  = T >> 8;
    // input flat index for (row n, channel ch): bb*262144 + ch*4096 + hh*64 + ww0 + m
    const int base = bb * 262144 + hh * 64 + ww0;

    // ---- A operands: 16x32 f16, layout: lanes 0-15 K∈{0-15}, lanes 16-31 K∈{16-31}
    v16h a0, a1;
    {
        const int rb = base + col;                  // this lane's row (M = col)
        #pragma unroll
        for (int t = 0; t < 16; ++t) {
            const int k = (g << 4) + t;  // 0..31
            a0[t] = (_Float16)inp[rb + k * 4096];
            a1[t] = (_Float16)inp[rb + (k + 32) * 4096];
        }
    }

    float best[8];
    int   bestd[8];
    #pragma unroll
    for (int j = 0; j < 8; ++j) { best[j] = -3.4e38f; bestd[j] = 0; }

    // ---- Sweep 32 d-tiles of 16 codes; score = f.e - 0.5*||e||^2 (monotone in ref score)
    for (int dt = 0; dt < 32; ++dt) {
        const int d = (dt << 4) + col;              // this lane's column code
        // B layout: lane = column, K = 16*g + t (contiguous halves in the code's row)
        const v8h* p0 = (const v8h*)(cb + d * EMB + (g << 4));
        const v8h* p1 = (const v8h*)(cb + d * EMB + 32 + (g << 4));
        v8h lo0 = p0[0], hi0 = p0[1];
        v8h lo1 = p1[0], hi1 = p1[1];
        v16h b0, b1;
        #pragma unroll
        for (int t = 0; t < 8; ++t) {
            b0[t] = lo0[t]; b0[t + 8] = hi0[t];
            b1[t] = lo1[t]; b1[t + 8] = hi1[t];
        }
        v8f c = {};
        c = __builtin_amdgcn_wmma_f32_16x16x32_f16(false, a0, false, b0, (short)0, c, false, false);
        c = __builtin_amdgcn_wmma_f32_16x16x32_f16(false, a1, false, b1, (short)0, c, false, false);
        const float hn = nrm[d];
        #pragma unroll
        for (int j = 0; j < 8; ++j) {
            const float s = c[j] - hn;              // row m = j + 8*g, code d
            if (s > best[j]) { best[j] = s; bestd[j] = d; }   // strict > keeps lowest d
        }
    }

    // ---- Cross-lane argmax over the 16 columns (lanes xor 1,2,4,8 stay in the half-wave)
    #pragma unroll
    for (int j = 0; j < 8; ++j) {
        float bv = best[j]; int bd = bestd[j];
        #pragma unroll
        for (int off = 1; off < 16; off <<= 1) {
            const float ov = __shfl_xor(bv, off, 32);
            const int   od = __shfl_xor(bd, off, 32);
            if (ov > bv || (ov == bv && od < bd)) { bv = ov; bd = od; }
        }
        best[j] = bv; bestd[j] = bd;
    }
    if (col == 0) {                                  // lanes 0 and 16
        #pragma unroll
        for (int j = 0; j < 8; ++j)
            idsh[wave * 16 + (g * 8 + j)] = bestd[j];
    }
    __syncthreads();

    // ---- Output phase: ids, quantized, commit loss, EMA statistics
    const int m    = col;                            // lane handles ww = ww0 + col
    const int myid = idsh[wave * 16 + m];
    const int n    = (bb * 64 + ww0 + m) * 64 + hh;  // flat row index (= ids flat order)
    if (g == 0) {
        ids_out[n] = (float)myid;
        atomicAdd(&ws_hist[myid], 1.0f);
    }
    float commit = 0.0f;
    #pragma unroll 4
    for (int it = 0; it < 32; ++it) {
        const int ch   = (it << 1) + g;              // g=0 even ch, g=1 odd ch
        const int addr = base + m + ch * 4096;       // lanes 0-15 contiguous -> coalesced
        const float x  = inp[addr];
        const float qv = embed[myid * EMB + ch];
        q_out[addr] = qv;
        const float diff = x - qv;
        commit += diff * diff;
        atomicAdd(&ws_esum[ch * DICT + myid], x);    // embed_sum[ch][d]
    }
    #pragma unroll
    for (int off = 16; off; off >>= 1)
        commit += __shfl_xor(commit, off, 32);
    if (lane == 0) atomicAdd(ws_commit, commit);
}

__global__ void vq_finalize(const float* __restrict__ cluster_size,  // (512,)
                            const float* __restrict__ embed_avg,     // (64, 512)
                            const float* __restrict__ ws_hist,
                            const float* __restrict__ ws_esum,
                            const float* __restrict__ ws_commit,
                            float* __restrict__ commit_out,
                            float* __restrict__ new_embed,           // (512, 64)
                            float* __restrict__ new_cs,              // (512,)
                            float* __restrict__ new_eavg)            // (64, 512)
{
    __shared__ float s_ncs[DICT];
    __shared__ float s_red[256];
    const int tid = threadIdx.x;
    float part = 0.0f;
    for (int d = tid; d < DICT; d += 256) {
        const float v = cluster_size[d] * 0.99f + ws_hist[d] * 0.01f;
        s_ncs[d] = v;
        part += v;
    }
    s_red[tid] = part;
    __syncthreads();
    for (int s = 128; s; s >>= 1) {
        if (tid < s) s_red[tid] += s_red[tid + s];
        __syncthreads();
    }
    const float n = s_red[0];
    if (tid == 0) commit_out[0] = ws_commit[0] * (1.0f / 8388608.0f);
    for (int d = tid; d < DICT; d += 256) new_cs[d] = s_ncs[d];
    const float denom = n + (float)DICT * 1e-5f;
    for (int i = tid; i < EMB * DICT; i += 256) {
        const int k = i >> 9;        // channel
        const int d = i & 511;       // code
        const float ea = embed_avg[i] * 0.99f + ws_esum[i] * 0.01f;
        new_eavg[i] = ea;
        const float cs = n * (s_ncs[d] + 1e-5f) / denom;
        new_embed[d * EMB + k] = ea / cs;
    }
}

extern "C" void kernel_launch(void* const* d_in, const int* in_sizes, int n_in,
                              void* d_out, int out_size, void* d_ws, size_t ws_size,
                              hipStream_t stream) {
    const float* inp          = (const float*)d_in[0];  // (32,64,64,64)
    const float* embed        = (const float*)d_in[1];  // (512,64)
    const float* cluster_size = (const float*)d_in[2];  // (512,)
    const float* embed_avg    = (const float*)d_in[3];  // (64,512)

    float* out        = (float*)d_out;
    float* q_out      = out;                 // 8388608
    float* commit_out = out + 8388608;       // 1
    float* ids_out    = out + 8388609;       // 131072
    float* new_embed  = out + 8519681;       // 32768
    float* new_cs     = out + 8552449;       // 512
    float* new_eavg   = out + 8552961;       // 32768

    float* ws        = (float*)d_ws;
    float* ws_hist   = ws;                   // 512
    float* ws_esum   = ws + 512;             // 64*512 = 32768
    float* ws_commit = ws + 512 + 32768;     // 1
    const int nz = 512 + 32768 + 1;

    vq_zero<<<(nz + 255) / 256, 256, 0, stream>>>(ws, nz);
    vq_main<<<1024, 256, LDS_TOTAL, stream>>>(inp, embed, q_out, ids_out,
                                              ws_hist, ws_esum, ws_commit);
    vq_finalize<<<1, 256, 0, stream>>>(cluster_size, embed_avg, ws_hist, ws_esum, ws_commit,
                                       commit_out, new_embed, new_cs, new_eavg);
}
